// myGRUCell_group4_54571854463571
// MI455X (gfx1250) — matrix operand order repaired and checked by reading, not compiled, not run on hardware
//
#include <hip/hip_runtime.h>
#include <hip/hip_bf16.h>
#include <stdint.h>

typedef __attribute__((ext_vector_type(16))) __bf16    v16bf;
typedef __attribute__((ext_vector_type(8)))  float     v8f;
typedef __attribute__((ext_vector_type(8)))  uint32_t  v8u;

#define NB     8192
#define HID    1024
#define INDIM  1024
#define WR     256
#define HG     256
#define TCOLS  768     // 4 groups * 192 concat rank

// ---- workspace layout (uint32 offsets) ----
#define OFF_WP     0u          // W packed      [512][256]   u32
#define OFF_W123   131072u     // W1,W2,W3 pk   3x[128][1024] u32
#define OFF_D0     524288u     // U    pk       [4][128][64]
#define OFF_D1     557056u     // UU   pk       [4][128][64]
#define OFF_D2     589824u     // UUU  pk       [4][128][32]
#define OFF_D3     606208u     // UUUU pk       [4][128][32]
#define OFF_V      622592u     // V1,V2,V3 pk   3x[4][96][256]
#define OFF_XBF    917504u     // x bf16 [8192][1024] as u32 [8192][512]
#define OFF_HBF    5111808u    // h bf16 [8192][1024] as u32 [8192][512]
#define OFF_XW     9306112u    // xw bf16 [8192][256] as u32 [8192][128]
#define OFF_T      10354688u   // T  bf16 [8192][768] as u32 [8192][384]
#define PREP_TOTAL 917504

__device__ __forceinline__ uint32_t pk2(float a, float b) {
  uint32_t ua = __builtin_bit_cast(uint32_t, a);
  uint32_t ub = __builtin_bit_cast(uint32_t, b);
  ua = (ua + 0x7FFFu + ((ua >> 16) & 1u)) >> 16;   // rne bf16
  ub = (ub + 0x7FFFu + ((ub >> 16) & 1u)) >> 16;
  return ua | (ub << 16);
}
__device__ __forceinline__ uint16_t bf1(float a) {
  uint32_t ua = __builtin_bit_cast(uint32_t, a);
  return (uint16_t)((ua + 0x7FFFu + ((ua >> 16) & 1u)) >> 16);
}

__device__ __forceinline__ v8f wmma_bf16(v8u a, v8u b, v8f c) {
  return __builtin_amdgcn_wmma_f32_16x16x32_bf16(
      false, __builtin_bit_cast(v16bf, a),
      false, __builtin_bit_cast(v16bf, b),
      (short)0, c, false, false);
}

// A-fragment K offset for VGPR j (16-bit A 16x32 layout, ISA 7.12.2)
__device__ __forceinline__ int ka_off(int j, int half) {
  return ((j < 4) ? (2 * j) : (16 + 2 * (j - 4))) + half * 8;
}
// B-fragment K offset for VGPR j (B 32x16 layout)
__device__ __forceinline__ int kb_off(int j, int half) {
  return ((j < 4) ? (2 * j) : (16 + 2 * (j - 4))) + half * 8;
}

// row k of packed per-group up-projection V_m[g] (concat U*,UU*,UUU*,UUUU*)
__device__ __forceinline__ float vrow(
    int m, int g, int k, int n,
    const float* U1, const float* U2, const float* U3,
    const float* UU1, const float* UU2, const float* UU3,
    const float* UUU1, const float* UUU2, const float* UUU3,
    const float* UUUU1, const float* UUUU2, const float* UUUU3) {
  if (k < 64)  { const float* M = (m == 0) ? U1 : ((m == 1) ? U2 : U3);
                 return M[(g * 64 + k) * HG + n]; }
  if (k < 128) { const float* M = (m == 0) ? UU1 : ((m == 1) ? UU2 : UU3);
                 return M[(g * 64 + (k - 64)) * HG + n]; }
  if (k < 160) { const float* M = (m == 0) ? UUU1 : ((m == 1) ? UUU2 : UUU3);
                 return M[(g * 32 + (k - 128)) * HG + n]; }
  const float* M = (m == 0) ? UUUU1 : ((m == 1) ? UUUU2 : UUUU3);
  return M[(g * 32 + (k - 160)) * HG + n];
}

// ---------------- kernel 0a: convert + pack all weights ----------------
__global__ __launch_bounds__(256) void k_prep(
    const float* __restrict__ W, const float* __restrict__ W1,
    const float* __restrict__ W2, const float* __restrict__ W3,
    const float* __restrict__ U, const float* __restrict__ UU,
    const float* __restrict__ UUU, const float* __restrict__ UUUU,
    const float* __restrict__ U1, const float* __restrict__ U2, const float* __restrict__ U3,
    const float* __restrict__ UU1, const float* __restrict__ UU2, const float* __restrict__ UU3,
    const float* __restrict__ UUU1, const float* __restrict__ UUU2, const float* __restrict__ UUU3,
    const float* __restrict__ UUUU1, const float* __restrict__ UUUU2, const float* __restrict__ UUUU3,
    uint32_t* __restrict__ ws) {
  int stride = gridDim.x * blockDim.x;
  for (int i = blockIdx.x * blockDim.x + threadIdx.x; i < PREP_TOTAL; i += stride) {
    if (i < 131072) {                 // W: [1024][256] -> pair-packed [512][256]
      int p = i >> 8, n = i & 255;
      ws[OFF_WP + i] = pk2(W[(2 * p) * WR + n], W[(2 * p + 1) * WR + n]);
    } else if (i < 524288) {          // W1/W2/W3: [256][1024] -> [128][1024]
      int j = i - 131072;
      int m = j >> 17;
      int rr = j & 131071;
      int p = rr >> 10, n = rr & 1023;
      const float* Wm = (m == 0) ? W1 : ((m == 1) ? W2 : W3);
      ws[OFF_W123 + j] = pk2(Wm[(2 * p) * HID + n], Wm[(2 * p + 1) * HID + n]);
    } else if (i < 622592) {          // down-projection sets
      int j = i - 524288;
      const float* src; int rdim; uint32_t offb; int jl;
      if (j < 32768)      { src = U;    rdim = 64; offb = OFF_D0; jl = j; }
      else if (j < 65536) { src = UU;   rdim = 64; offb = OFF_D1; jl = j - 32768; }
      else if (j < 81920) { src = UUU;  rdim = 32; offb = OFF_D2; jl = j - 65536; }
      else                { src = UUUU; rdim = 32; offb = OFF_D3; jl = j - 81920; }
      int per_g = 128 * rdim;
      int g = jl / per_g;
      int rm = jl - g * per_g;
      int p = rm / rdim;
      int n = rm - p * rdim;
      ws[offb + jl] = pk2(src[(g * HG + 2 * p) * rdim + n],
                          src[(g * HG + 2 * p + 1) * rdim + n]);
    } else {                          // packed V1/V2/V3: [4][192][256] -> [4][96][256]
      int j = i - 622592;
      int m = j / 98304;
      int r = j - m * 98304;
      int g = r / (96 * 256);
      int rr = r - g * (96 * 256);
      int p = rr >> 8, n = rr & 255;
      float a = vrow(m, g, 2 * p,     n, U1, U2, U3, UU1, UU2, UU3,
                     UUU1, UUU2, UUU3, UUUU1, UUUU2, UUUU3);
      float b = vrow(m, g, 2 * p + 1, n, U1, U2, U3, UU1, UU2, UU3,
                     UUU1, UUU2, UUU3, UUUU1, UUUU2, UUUU3);
      ws[OFF_V + j] = pk2(a, b);
    }
  }
}

// ---------------- kernel 0b: stage x/h as row-major bf16 pairs ----------------
__global__ __launch_bounds__(256) void k_cvt(const float* __restrict__ x,
                                             const float* __restrict__ h,
                                             uint32_t* __restrict__ ws) {
  const int total = NB * 512;                    // u32 pairs per tensor
  int stride = gridDim.x * blockDim.x;
  const float2* x2 = (const float2*)x;
  const float2* h2 = (const float2*)h;
  for (int i = blockIdx.x * blockDim.x + threadIdx.x; i < total; i += stride) {
    float2 xv = x2[i];
    ws[OFF_XBF + i] = pk2(xv.x, xv.y);
    float2 hv = h2[i];
    ws[OFF_HBF + i] = pk2(hv.x, hv.y);
  }
}

// ------- kernel 1: xw = bf16(x @ W): 16x64 tile per wave, 4 accums -------
__global__ __launch_bounds__(256) void k_gemm_xw(uint32_t* __restrict__ ws) {
  const uint32_t* xbf = ws + OFF_XBF;            // [8192][512] u32
  const uint32_t* Wp  = ws + OFF_WP;             // [512][256]  u32
  uint16_t* xwbf = (uint16_t*)(ws + OFF_XW);
  int lane = threadIdx.x & 31, wave = threadIdx.x >> 5;
  int tile = blockIdx.x * 8 + wave;              // 512 * 4 jobs
  int tm = tile >> 2, tj = tile & 3;
  int row0 = tm * 16, col0 = tj * 64;
  int half = lane >> 4, rl = lane & 15;
  int arow = row0 + rl;
  v8f acc[4];
#pragma unroll
  for (int t = 0; t < 4; ++t) acc[t] = {};
  for (int kb = 0; kb < INDIM; kb += 32) {
    v8u a;
#pragma unroll
    for (int j = 0; j < 8; ++j)
      a[j] = xbf[arow * 512 + ((kb + ka_off(j, half)) >> 1)];
    // unconditional speculative prefetch of next A block (stays inside ws)
    __builtin_prefetch(&xbf[arow * 512 + ((kb + 32) >> 1)], 0, 3);
#pragma unroll
    for (int t = 0; t < 4; ++t) {
      v8u b;
#pragma unroll
      for (int j = 0; j < 8; ++j) {
        int p = (kb + kb_off(j, half)) >> 1;
        b[j] = Wp[p * WR + col0 + t * 16 + rl];
      }
      acc[t] = wmma_bf16(a, b, acc[t]);
    }
  }
#pragma unroll
  for (int t = 0; t < 4; ++t)
#pragma unroll
    for (int v = 0; v < 8; ++v) {
      int r = row0 + v + half * 8;
      xwbf[r * WR + col0 + t * 16 + rl] = bf1(acc[t][v]);
    }
}

// ------- kernel 2: grouped down-projections, one (g,set) segment/wave -------
template <int NT>
__device__ __forceinline__ void gemm_t_body(const uint32_t* __restrict__ hbf,
                                            const uint32_t* __restrict__ base,
                                            uint16_t* __restrict__ Tbf,
                                            int row0, int colT, int s,
                                            int half, int rl) {
  constexpr int RD = NT * 16;
  int arow = row0 + rl;
  v8f acc[NT];
#pragma unroll
  for (int t = 0; t < NT; ++t) acc[t] = {};
  for (int kb = 0; kb < HG; kb += 32) {
    v8u a;
#pragma unroll
    for (int j = 0; j < 8; ++j)
      a[j] = hbf[arow * 512 + s * 128 + ((kb + ka_off(j, half)) >> 1)];
#pragma unroll
    for (int t = 0; t < NT; ++t) {
      v8u b;
#pragma unroll
      for (int j = 0; j < 8; ++j) {
        int p = (kb + kb_off(j, half)) >> 1;
        b[j] = base[p * RD + t * 16 + rl];
      }
      acc[t] = wmma_bf16(a, b, acc[t]);
    }
  }
#pragma unroll
  for (int t = 0; t < NT; ++t)
#pragma unroll
    for (int v = 0; v < 8; ++v) {
      int r = row0 + v + half * 8;
      Tbf[r * TCOLS + colT + t * 16 + rl] = bf1(acc[t][v]);
    }
}

__global__ __launch_bounds__(256) void k_gemm_t(uint32_t* __restrict__ ws) {
  const uint32_t* hbf = ws + OFF_HBF;            // [8192][512] u32
  uint16_t* Tbf = (uint16_t*)(ws + OFF_T);
  int lane = threadIdx.x & 31, wave = threadIdx.x >> 5;
  int tile = blockIdx.x * 8 + wave;              // 512 * 16 jobs
  int tm = tile >> 4, js = tile & 15;
  int g = js >> 2, set = js & 3;
  int row0 = tm * 16;
  int s = (g + set) & 3;                         // roll(-shift)
  int half = lane >> 4, rl = lane & 15;
  if (set == 0) {
    gemm_t_body<4>(hbf, ws + OFF_D0 + g * 128 * 64, Tbf, row0, g * 192 + 0,   s, half, rl);
  } else if (set == 1) {
    gemm_t_body<4>(hbf, ws + OFF_D1 + g * 128 * 64, Tbf, row0, g * 192 + 64,  s, half, rl);
  } else if (set == 2) {
    gemm_t_body<2>(hbf, ws + OFF_D2 + g * 128 * 32, Tbf, row0, g * 192 + 128, s, half, rl);
  } else {
    gemm_t_body<2>(hbf, ws + OFF_D3 + g * 128 * 32, Tbf, row0, g * 192 + 160, s, half, rl);
  }
}

// ---- kernel 3: fused stage-2 GEMMs (r,z,w3,u3sum) + GRU epilogue, 16x32/wave ----
__global__ __launch_bounds__(256) void k_fused(const float* __restrict__ h,
                                               const float* __restrict__ br,
                                               const float* __restrict__ bg,
                                               const float* __restrict__ bu,
                                               const uint32_t* __restrict__ ws,
                                               float* __restrict__ out) {
  int lane = threadIdx.x & 31, wave = threadIdx.x >> 5;
  int tile = blockIdx.x * 8 + wave;              // 512 * 32 jobs
  int tm = tile >> 5, tn = tile & 31;
  int row0 = tm * 16, col0 = tn * 32;            // two 16-wide subtiles
  int g = col0 >> 8;
  int nl = col0 & 255;
  int half = lane >> 4, rl = lane & 15;
  int arow = row0 + rl;
  const uint32_t* xw  = ws + OFF_XW;             // [8192][128] u32
  const uint32_t* Tm  = ws + OFF_T;              // [8192][384] u32
  const uint32_t* W1p = ws + OFF_W123;
  const uint32_t* W2p = W1p + 131072;
  const uint32_t* W3p = W2p + 131072;
  const uint32_t* V1p = ws + OFF_V + g * 96 * 256;
  const uint32_t* V2p = V1p + 98304;
  const uint32_t* V3p = V2p + 98304;
  v8f aR[2], aZ[2], aW[2], aU[2];
#pragma unroll
  for (int t = 0; t < 2; ++t) { aR[t] = {}; aZ[t] = {}; aW[t] = {}; aU[t] = {}; }
  // stage-2a: xw (K=256) vs W1/W2/W3
  for (int kb = 0; kb < WR; kb += 32) {
    v8u a;
#pragma unroll
    for (int j = 0; j < 8; ++j)
      a[j] = xw[arow * 128 + ((kb + ka_off(j, half)) >> 1)];
#pragma unroll
    for (int t = 0; t < 2; ++t) {
      int c = col0 + t * 16 + rl;
      v8u b1, b2, b3;
#pragma unroll
      for (int j = 0; j < 8; ++j) {
        int p = (kb + kb_off(j, half)) >> 1;
        b1[j] = W1p[p * HID + c];
        b2[j] = W2p[p * HID + c];
        b3[j] = W3p[p * HID + c];
      }
      aR[t] = wmma_bf16(a, b1, aR[t]);
      aZ[t] = wmma_bf16(a, b2, aZ[t]);
      aW[t] = wmma_bf16(a, b3, aW[t]);
    }
  }
  // stage-2b: T group block (K=192) vs packed V1/V2/V3
  for (int kb = 0; kb < 192; kb += 32) {
    v8u a;
#pragma unroll
    for (int j = 0; j < 8; ++j)
      a[j] = Tm[arow * 384 + ((g * 192 + kb + ka_off(j, half)) >> 1)];
#pragma unroll
    for (int t = 0; t < 2; ++t) {
      int c = nl + t * 16 + rl;
      v8u b1, b2, b3;
#pragma unroll
      for (int j = 0; j < 8; ++j) {
        int p = (kb + kb_off(j, half)) >> 1;
        b1[j] = V1p[p * HG + c];
        b2[j] = V2p[p * HG + c];
        b3[j] = V3p[p * HG + c];
      }
      aR[t] = wmma_bf16(a, b1, aR[t]);
      aZ[t] = wmma_bf16(a, b2, aZ[t]);
      aU[t] = wmma_bf16(a, b3, aU[t]);
    }
  }
#pragma unroll
  for (int t = 0; t < 2; ++t) {
    int col = col0 + t * 16 + rl;
    float vbr = br[col], vbg = bg[col], vbu = bu[col];
#pragma unroll
    for (int v = 0; v < 8; ++v) {
      int r = row0 + v + half * 8;
      float rr = 1.0f / (1.0f + __expf(-(aR[t][v] + vbr)));
      float zz = 1.0f / (1.0f + __expf(-(aZ[t][v] + vbg)));
      float cc = tanhf(aW[t][v] + rr * aU[t][v] + vbu);
      float hv = h[r * HID + col];
      out[r * HID + col] = zz * hv + (1.0f - zz) * cc;
    }
  }
}

extern "C" void kernel_launch(void* const* d_in, const int* in_sizes, int n_in,
                              void* d_out, int out_size, void* d_ws, size_t ws_size,
                              hipStream_t stream) {
  const float* x     = (const float*)d_in[0];
  const float* h     = (const float*)d_in[1];
  const float* W     = (const float*)d_in[2];
  const float* W1    = (const float*)d_in[3];
  const float* W2    = (const float*)d_in[4];
  const float* W3    = (const float*)d_in[5];
  const float* U     = (const float*)d_in[6];
  const float* U1    = (const float*)d_in[7];
  const float* U2    = (const float*)d_in[8];
  const float* U3    = (const float*)d_in[9];
  const float* UU    = (const float*)d_in[10];
  const float* UU1   = (const float*)d_in[11];
  const float* UU2   = (const float*)d_in[12];
  const float* UU3   = (const float*)d_in[13];
  const float* UUU   = (const float*)d_in[14];
  const float* UUU1  = (const float*)d_in[15];
  const float* UUU2  = (const float*)d_in[16];
  const float* UUU3  = (const float*)d_in[17];
  const float* UUUU  = (const float*)d_in[18];
  const float* UUUU1 = (const float*)d_in[19];
  const float* UUUU2 = (const float*)d_in[20];
  const float* UUUU3 = (const float*)d_in[21];
  const float* br    = (const float*)d_in[22];
  const float* bg    = (const float*)d_in[23];
  const float* bu    = (const float*)d_in[24];
  uint32_t* ws = (uint32_t*)d_ws;
  float* out = (float*)d_out;

  k_prep<<<512, 256, 0, stream>>>(W, W1, W2, W3, U, UU, UUU, UUUU,
                                  U1, U2, U3, UU1, UU2, UU3,
                                  UUU1, UUU2, UUU3, UUUU1, UUUU2, UUUU3, ws);
  k_cvt<<<2048, 256, 0, stream>>>(x, h, ws);
  k_gemm_xw<<<256, 256, 0, stream>>>(ws);        // 512*4 jobs / 8 waves
  k_gemm_t<<<1024, 256, 0, stream>>>(ws);        // 512*16 jobs / 8 waves
  k_fused<<<2048, 256, 0, stream>>>(h, br, bg, bu, ws, out);  // 512*32 jobs
}
